// KeyChannelwiseMemoryMultiHead_4080218931760
// MI455X (gfx1250) — matrix operand order repaired and checked by reading, not compiled, not run on hardware
//
#include <hip/hip_runtime.h>
#include <hip/hip_bf16.h>

typedef unsigned short u16;
typedef unsigned int   u32;
typedef __attribute__((ext_vector_type(16))) __bf16 v16bf;
typedef __attribute__((ext_vector_type(8)))  __bf16 v8bf;
typedef __attribute__((ext_vector_type(8)))  float  v8f;

static __device__ __forceinline__ u16 f32_to_bf16(float f) {
    u32 u = __builtin_bit_cast(u32, f);
    u += 0x7FFFu + ((u >> 16) & 1u);          // round-to-nearest-even
    return (u16)(u >> 16);
}

// A-matrix 16-bit fragment per CDNA5 ISA layout (16x32, MxK):
// lane half h = lane>>4 : elements 0..7 = K[k0+8h .. +7], elements 8..15 = K[k0+16+8h .. +7]
static __device__ __forceinline__ v16bf load_afrag(const u16* __restrict__ base, int elemOffLo) {
    v8bf lo = *reinterpret_cast<const v8bf*>(base + elemOffLo);
    v8bf hi = *reinterpret_cast<const v8bf*>(base + elemOffLo + 16);
    v16bf r;
#pragma unroll
    for (int i = 0; i < 8; ++i) { r[i] = lo[i]; r[8 + i] = hi[i]; }
    return r;
}

static __device__ __forceinline__ v8f wmma_bf16(v16bf a, v16bf b, v8f c) {
    return __builtin_amdgcn_wmma_f32_16x16x32_bf16(
        /*neg_a=*/false, a, /*neg_b=*/false, b,
        /*c_mod=*/(short)0, c, /*reuse_a=*/false, /*reuse_b=*/false);
}

// ---------------- weight-fusion precompute ----------------
// W1[n*64+m][c] = sum_k key_p[n,k,m] * w_in[k*8+n, c]   (bf16, [512,256])
// b1[n*64+m]    = sum_k key_p[n,k,m] * b_in[k*8+n]      (f32,  [512])
__global__ __launch_bounds__(256) void build_w1(
    const float* __restrict__ key_p, const float* __restrict__ w_in,
    const float* __restrict__ b_in, u16* __restrict__ W1, float* __restrict__ b1)
{
    const int r = blockIdx.x;           // 0..511
    const int c = threadIdx.x;          // 0..255
    const int n = r >> 6, m = r & 63;
    float s = 0.f;
#pragma unroll 8
    for (int k = 0; k < 64; ++k)
        s += key_p[n * 4096 + k * 64 + m] * w_in[(k * 8 + n) * 256 + c];
    W1[r * 256 + c] = f32_to_bf16(s);
    if (c == 0) {
        float sb = 0.f;
#pragma unroll 8
        for (int k = 0; k < 64; ++k)
            sb += key_p[n * 4096 + k * 64 + m] * b_in[k * 8 + n];
        b1[r] = sb;
    }
}

// W2[o][n*64+m] = sum_d w_out[o, n*64+d] * memory[n,m,d]   (bf16, [256,512])
__global__ __launch_bounds__(512) void build_w2(
    const float* __restrict__ w_out, const float* __restrict__ memory,
    u16* __restrict__ W2)
{
    const int o = blockIdx.x;           // 0..255
    const int r = threadIdx.x;          // 0..511
    const int n = r >> 6, m = r & 63;
    float s = 0.f;
#pragma unroll 8
    for (int d = 0; d < 64; ++d)
        s += w_out[o * 512 + n * 64 + d] * memory[n * 4096 + m * 64 + d];
    W2[o * 512 + r] = f32_to_bf16(s);
}

// ---------------- fused main kernel ----------------
#define TPIX 32          // pixels per tile
#define XSTR 272         // 256 + 16 pad (keeps 32B row alignment, breaks bank patterns)
#define PSTR 528         // 512 + 16 pad

__global__ __launch_bounds__(256) void fused_attention(
    const float* __restrict__ x,      // [8,256,64,64]
    const u16*   __restrict__ W1,     // [512,256] bf16
    const float* __restrict__ b1,     // [512]
    const u16*   __restrict__ W2,     // [256,512] bf16
    const float* __restrict__ b_out,  // [256]
    float* __restrict__ out)          // [8,256,64,64]
{
    __shared__ u16 Xs[TPIX * XSTR];   // x tile, bf16, [pix][c]
    __shared__ u16 P [TPIX * PSTR];   // softmax probs, bf16, [pix][n*64+m]

    const int tile = blockIdx.x;             // 1024 tiles
    const int p0   = tile * TPIX;
    const int b    = p0 >> 12;               // batch (4096 px per image)
    const int s0   = p0 & 4095;              // spatial offset within image
    const int wave = threadIdx.x >> 5;       // 0..7
    const int lane = threadIdx.x & 31;
    const int half = lane >> 4;              // K-half selector
    const int l16  = lane & 15;              // M/N index within fragment

    // ---- stage 0: load X tile (coalesced 128B/wave), convert to bf16 ----
    {
        const float* xb = x + (size_t)b * (256 * 4096) + s0 + lane;
#pragma unroll
        for (int i = 0; i < 32; ++i) {
            const int c = wave + i * 8;
            Xs[lane * XSTR + c] = f32_to_bf16(xb[(size_t)c * 4096]);
        }
    }
    __syncthreads();

    // ---- stage 1: S[head*64+m, pix] = W1 @ X + b1 ; wave 'w' owns head w ----
    const int head  = wave;
    const int rbase = head * 64;
    v8f acc[4][2];
#pragma unroll
    for (int mt = 0; mt < 4; ++mt)
#pragma unroll
        for (int nb = 0; nb < 2; ++nb) acc[mt][nb] = (v8f)(0.f);

#pragma unroll
    for (int kb = 0; kb < 8; ++kb) {
        const int k0 = kb * 32;
        v16bf Bf[2];
#pragma unroll
        for (int nb = 0; nb < 2; ++nb) {
            const int pix = nb * 16 + l16;
            Bf[nb] = load_afrag(Xs, pix * XSTR + k0 + 8 * half);
        }
#pragma unroll
        for (int mt = 0; mt < 4; ++mt) {
            const int row = rbase + mt * 16 + l16;
            const v16bf Af = load_afrag(W1, row * 256 + k0 + 8 * half);
#pragma unroll
            for (int nb = 0; nb < 2; ++nb)
                acc[mt][nb] = wmma_bf16(Af, Bf[nb], acc[mt][nb]);
        }
    }

    // ---- stage 2: bias + in-register softmax over the head's 64 m-rows ----
#pragma unroll
    for (int mt = 0; mt < 4; ++mt)
#pragma unroll
        for (int v = 0; v < 8; ++v) {
            const float bb = b1[rbase + mt * 16 + 8 * half + v];
            acc[mt][0][v] += bb;
            acc[mt][1][v] += bb;
        }

#pragma unroll
    for (int nb = 0; nb < 2; ++nb) {
        const int pix = nb * 16 + l16;
        float mx = -3.0e38f;
#pragma unroll
        for (int mt = 0; mt < 4; ++mt)
#pragma unroll
            for (int v = 0; v < 8; ++v) mx = fmaxf(mx, acc[mt][nb][v]);
        mx = fmaxf(mx, __shfl_xor(mx, 16));       // combine lane L with L^16
        float sum = 0.f;
#pragma unroll
        for (int mt = 0; mt < 4; ++mt)
#pragma unroll
            for (int v = 0; v < 8; ++v) {
                const float e = __expf(acc[mt][nb][v] - mx);
                acc[mt][nb][v] = e;
                sum += e;
            }
        sum += __shfl_xor(sum, 16);
        const float inv = 1.0f / sum;
#pragma unroll
        for (int mt = 0; mt < 4; ++mt)
#pragma unroll
            for (int v = 0; v < 8; v += 2) {
                const u32 packed =
                    (u32)f32_to_bf16(acc[mt][nb][v] * inv) |
                    ((u32)f32_to_bf16(acc[mt][nb][v + 1] * inv) << 16);
                *reinterpret_cast<u32*>(&P[pix * PSTR + rbase + mt * 16 + 8 * half + v]) = packed;
            }
    }
    __syncthreads();

    // ---- stage 3: Y[256, pix] = W2 @ P ; wave 'w' owns output rows 32w..32w+31 ----
    v8f acc2[2][2];
#pragma unroll
    for (int i = 0; i < 2; ++i)
#pragma unroll
        for (int nb = 0; nb < 2; ++nb) acc2[i][nb] = (v8f)(0.f);

#pragma unroll 4
    for (int kb = 0; kb < 16; ++kb) {
        const int k0 = kb * 32;
        v16bf Pf[2];
#pragma unroll
        for (int nb = 0; nb < 2; ++nb) {
            const int pix = nb * 16 + l16;
            Pf[nb] = load_afrag(P, pix * PSTR + k0 + 8 * half);
        }
#pragma unroll
        for (int i = 0; i < 2; ++i) {
            const int row = (wave * 2 + i) * 16 + l16;
            const v16bf Af = load_afrag(W2, row * 512 + k0 + 8 * half);
#pragma unroll
            for (int nb = 0; nb < 2; ++nb)
                acc2[i][nb] = wmma_bf16(Af, Pf[nb], acc2[i][nb]);
        }
    }

    // ---- epilogue: +b_out, store NCHW f32 (two 64B bursts per store op) ----
    float* ob = out + (size_t)b * (256 * 4096) + s0;
#pragma unroll
    for (int i = 0; i < 2; ++i)
#pragma unroll
        for (int v = 0; v < 8; ++v) {
            const int o  = (wave * 2 + i) * 16 + 8 * half + v;
            const float bo = b_out[o];
#pragma unroll
            for (int nb = 0; nb < 2; ++nb) {
                const int col = nb * 16 + l16;
                ob[(size_t)o * 4096 + col] = acc2[i][nb][v] + bo;
            }
        }
}

extern "C" void kernel_launch(void* const* d_in, const int* in_sizes, int n_in,
                              void* d_out, int out_size, void* d_ws, size_t ws_size,
                              hipStream_t stream) {
    const float* x      = (const float*)d_in[0];  // [8,256,64,64]
    const float* key_p  = (const float*)d_in[1];  // [8,64,64]
    const float* memory = (const float*)d_in[2];  // [8,64,64]
    const float* w_in   = (const float*)d_in[3];  // [512,256]
    const float* b_in   = (const float*)d_in[4];  // [512]
    const float* w_out  = (const float*)d_in[5];  // [256,512]
    const float* b_out  = (const float*)d_in[6];  // [256]
    float* out = (float*)d_out;

    char* ws = (char*)d_ws;
    u16*   W1 = (u16*)(ws);                        // 512*256*2 = 256 KB
    u16*   W2 = (u16*)(ws + 512 * 256 * 2);        // 256*512*2 = 256 KB
    float* b1 = (float*)(ws + 2 * 512 * 256 * 2);  // 2 KB

    build_w1<<<512, 256, 0, stream>>>(key_p, w_in, b_in, W1, b1);
    build_w2<<<256, 512, 0, stream>>>(w_out, memory, W2);
    fused_attention<<<(8 * 64 * 64) / TPIX, 256, 0, stream>>>(x, W1, b1, W2, b_out, out);
}
